// Warper_71657234366851
// MI455X (gfx1250) — compile-verified, matrix operand order checked by reading, not run on hardware
//
#include <hip/hip_runtime.h>
#include <hip/hip_bf16.h>

#define Zc 128
#define Hc 256
#define Sc 4

typedef __attribute__((ext_vector_type(16))) __bf16 v16bf;
typedef __attribute__((ext_vector_type(8)))  float  v8f;
typedef __attribute__((ext_vector_type(4)))  unsigned int v4u;

union Frag {
  v16bf v;
  unsigned short u[16];
  v4u q[2];
};

__device__ __forceinline__ unsigned short f2bf(float f){
  unsigned int u = __float_as_uint(f);
  u += 0x7FFFu + ((u >> 16) & 1u);      // round-to-nearest-even
  return (unsigned short)(u >> 16);
}
__device__ __forceinline__ float bf2f(unsigned short h){
  return __uint_as_float(((unsigned int)h) << 16);
}

// Branch-free tanh: prefer the gfx1250 V_TANH_F32 transcendental.
__device__ __forceinline__ float fast_tanh(float x){
#if __has_builtin(__builtin_amdgcn_tanhf)
  return __builtin_amdgcn_tanhf(x);
#else
  // 1 - 2/(e^{2x}+1); saturates correctly at +/-inf, no EXEC branching.
  float e = __expf(2.0f * x);
  return 1.0f - 2.0f * __builtin_amdgcn_rcpf(e + 1.0f);
#endif
}

// LDS layout (half-indices for weight fragment pool, byte offsets after)
#define OFF_WC 0               // 4*16 frags * 512 halves
#define OFF_W1 32768           // 1*16 frags
#define OFF_W2 40960           // 8*16 frags
#define OFF_W4 106496          // 8*1 frags
#define OFF_BIAS_B 221184      // bytes: bc[256] b1[256] b2[256] b4[16] f32
#define OFF_STAGE_B 224320     // bytes: 8 waves * 4096 halves (8KB) psf staging
#define OFF_K_B    (224320 + 65536)  // bytes: 8 waves * 64 floats k-exchange
#define LDS_BYTES  (OFF_K_B + 2048)

// Convert a row-major (Kreal x Nreal) f32 weight into bf16 B-fragments in LDS.
// Fragment (kt,nt) is 512 halves; lane l holds 16 halves; element e maps to
// K_in = (e&7) | (g<<3) | ((e&8)<<1) with g = lane>>4, n = nt*16 + (lane&15).
__device__ __forceinline__ void load_frags(const float* __restrict__ W,
                                           int Kreal, int Nreal, int KT, int NT,
                                           unsigned short* dst, int tid){
  int total = KT * NT * 512;
  for (int idx = tid; idx < total; idx += 256){
    int f  = idx >> 9;
    int w  = idx & 511;
    int ln = w >> 4;
    int e  = w & 15;
    int nt = f % NT;
    int kt = f / NT;
    int g  = ln >> 4;
    int nl = ln & 15;
    int Kin = (e & 7) | (g << 3) | ((e & 8) << 1);
    int K = kt * 32 + Kin;
    int n = nt * 16 + nl;
    float v = 0.0f;
    if (K < Kreal && n < Nreal) v = W[K * Nreal + n];
    dst[idx] = f2bf(v);
  }
}

__device__ __forceinline__ v8f wmma_bf16(v16bf a, const unsigned short* bp, int lane, v8f c){
  Frag b;
  b.q[0] = *(const v4u*)(bp + lane * 16);
  b.q[1] = *(const v4u*)(bp + lane * 16 + 8);
  return __builtin_amdgcn_wmma_f32_16x16x32_bf16(false, a, false, b.v, (short)0, c, false, false);
}

__global__ __launch_bounds__(256, 1) void warper_kernel(
    const float* __restrict__ input,
    const float* __restrict__ W1, const float* __restrict__ b1,
    const float* __restrict__ W2, const float* __restrict__ b2,
    const float* __restrict__ W4, const float* __restrict__ b4,
    const float* __restrict__ Wc, const float* __restrict__ bc,
    float* __restrict__ out, int N)
{
  extern __shared__ char smem[];
  unsigned short* lw = (unsigned short*)smem;
  float* lb = (float*)(smem + OFF_BIAS_B);

  const int tid  = threadIdx.x;
  const int lane = tid & 31;
  const int wv   = tid >> 5;
  const int g    = lane >> 4;
  const int ml   = lane & 15;

  unsigned short* st = (unsigned short*)(smem + OFF_STAGE_B) + wv * 4096;
  float* kw = (float*)(smem + OFF_K_B) + wv * 64;

  const int base = blockIdx.x * 128 + wv * 16;
  const int p  = base + ml;
  const int pc = p < N ? p : (N - 1);
  const float* row = input + (long long)pc * (Zc + 3);

  float x = row[Zc + 0], y = row[Zc + 1], z = row[Zc + 2];
  const float h = 1.0f / (float)Sc;
  const long long N3 = (long long)N * 3;

  #pragma unroll 1
  for (int s = 0; s < Sc; ++s){
    __syncthreads();
    load_frags(Wc + s*Zc*Hc, Zc, Hc, 4, 16, lw + OFF_WC, tid);
    load_frags(W1 + s*3*Hc,  3,  Hc, 1, 16, lw + OFF_W1, tid);
    load_frags(W2 + s*Hc*Hc, Hc, Hc, 8, 16, lw + OFF_W2, tid);
    load_frags(W4 + s*Hc*3,  Hc, 3,  8, 1,  lw + OFF_W4, tid);
    for (int i = tid; i < Hc; i += 256){
      lb[i]       = bc[s*Hc + i];
      lb[256 + i] = b1[s*Hc + i];
      lb[512 + i] = b2[s*Hc + i];
    }
    if (tid < 16) lb[768 + tid] = (tid < 3) ? b4[s*3 + tid] : 0.0f;
    __syncthreads();

    // ---- sf = tanh(code @ Wc + bc), held packed bf16 in 64 VGPRs/wave ----
    Frag ac[4];
    #pragma unroll
    for (int kt = 0; kt < 4; ++kt){
      #pragma unroll
      for (int e = 0; e < 16; ++e){
        int Kin = (e & 7) | (g << 3) | ((e & 8) << 1);
        ac[kt].u[e] = f2bf(row[kt*32 + Kin]);
      }
    }
    unsigned int sfp[16][4];
    #pragma unroll
    for (int nt = 0; nt < 16; ++nt){
      v8f acc = {};
      #pragma unroll
      for (int kt = 0; kt < 4; ++kt)
        acc = wmma_bf16(ac[kt].v, lw + OFF_WC + (kt*16 + nt)*512, lane, acc);
      float bias = lb[nt*16 + ml];
      #pragma unroll
      for (int k = 0; k < 4; ++k){
        float t0 = fast_tanh(acc[2*k + 0] + bias);
        float t1 = fast_tanh(acc[2*k + 1] + bias);
        sfp[nt][k] = (unsigned int)f2bf(t0) | ((unsigned int)f2bf(t1) << 16);
      }
    }

    // ---- one MLP evaluation f(x) ----
    auto feval = [&](float fx, float fy, float fz, float& kx, float& ky, float& kz){
      Frag ax;
      #pragma unroll
      for (int e = 0; e < 16; ++e) ax.u[e] = 0;
      if (g == 0){ ax.u[0] = f2bf(fx); ax.u[1] = f2bf(fy); ax.u[2] = f2bf(fz); }

      // pf = relu(x@W1 + b1); psf = pf * sf   (kept packed bf16: 64 VGPRs)
      unsigned int psf_pk[16][4];
      #pragma unroll
      for (int nt = 0; nt < 16; ++nt){
        v8f acc = {};
        acc = wmma_bf16(ax.v, lw + OFF_W1 + nt*512, lane, acc);
        float bias = lb[256 + nt*16 + ml];
        #pragma unroll
        for (int k = 0; k < 4; ++k){
          float p0 = acc[2*k + 0] + bias;  p0 = p0 > 0.0f ? p0 : 0.0f;
          float p1 = acc[2*k + 1] + bias;  p1 = p1 > 0.0f ? p1 : 0.0f;
          unsigned int pr = sfp[nt][k];
          p0 *= bf2f((unsigned short)(pr & 0xFFFF));
          p1 *= bf2f((unsigned short)(pr >> 16));
          psf_pk[nt][k] = (unsigned int)f2bf(p0) | ((unsigned int)f2bf(p1) << 16);
        }
      }
      // stage psf as bf16 [m][K] so A-fragments read as 2x ds_load_b128
      #pragma unroll
      for (int nt = 0; nt < 16; ++nt){
        #pragma unroll
        for (int k = 0; k < 4; ++k){
          unsigned int pr = psf_pk[nt][k];
          st[(2*k + 0 + 8*g)*256 + nt*16 + ml] = (unsigned short)(pr & 0xFFFF);
          st[(2*k + 1 + 8*g)*256 + nt*16 + ml] = (unsigned short)(pr >> 16);
        }
      }
      __syncthreads();
      Frag ap[8];
      #pragma unroll
      for (int kt = 0; kt < 8; ++kt){
        int r1 = ml*256 + kt*32 + 8*g;
        ap[kt].q[0] = *(const v4u*)(st + r1);
        ap[kt].q[1] = *(const v4u*)(st + r1 + 16);
      }
      __syncthreads();

      // psf2 = relu(psf@W2 + b2) + psf, restaged for W4
      #pragma unroll
      for (int nt = 0; nt < 16; ++nt){
        v8f acc = {};
        #pragma unroll
        for (int kt = 0; kt < 8; ++kt)
          acc = wmma_bf16(ap[kt].v, lw + OFF_W2 + (kt*16 + nt)*512, lane, acc);
        float bias = lb[512 + nt*16 + ml];
        #pragma unroll
        for (int k = 0; k < 4; ++k){
          unsigned int pr = psf_pk[nt][k];
          float o0 = acc[2*k + 0] + bias;  o0 = o0 > 0.0f ? o0 : 0.0f;
          float o1 = acc[2*k + 1] + bias;  o1 = o1 > 0.0f ? o1 : 0.0f;
          o0 += bf2f((unsigned short)(pr & 0xFFFF));
          o1 += bf2f((unsigned short)(pr >> 16));
          st[(2*k + 0 + 8*g)*256 + nt*16 + ml] = f2bf(o0);
          st[(2*k + 1 + 8*g)*256 + nt*16 + ml] = f2bf(o1);
        }
      }
      __syncthreads();
      Frag a2[8];
      #pragma unroll
      for (int kt = 0; kt < 8; ++kt){
        int r1 = ml*256 + kt*32 + 8*g;
        a2[kt].q[0] = *(const v4u*)(st + r1);
        a2[kt].q[1] = *(const v4u*)(st + r1 + 16);
      }
      v8f acc4 = {};
      #pragma unroll
      for (int kt = 0; kt < 8; ++kt)
        acc4 = wmma_bf16(a2[kt].v, lw + OFF_W4 + kt*512, lane, acc4);
      __syncthreads();

      // k = tanh(psf2@W4 + b4); redistribute 16x3 across lanes via LDS
      float bias4 = lb[768 + ml];
      if (ml < 3){
        #pragma unroll
        for (int v = 0; v < 8; ++v)
          kw[(v + 8*g)*4 + ml] = fast_tanh(acc4[v] + bias4);
      }
      __syncthreads();
      kx = kw[ml*4 + 0];
      ky = kw[ml*4 + 1];
      kz = kw[ml*4 + 2];
      __syncthreads();
    };

    // ---- RK4 ----
    float kx = 0.0f, ky = 0.0f, kz = 0.0f;
    float sx = 0.0f, sy = 0.0f, sz = 0.0f;
    const float cin[4] = {0.0f, 0.5f*h, 0.5f*h, h};
    const float cwt[4] = {1.0f, 2.0f, 2.0f, 1.0f};
    #pragma unroll 1
    for (int r = 0; r < 4; ++r){
      feval(x + cin[r]*kx, y + cin[r]*ky, z + cin[r]*kz, kx, ky, kz);
      sx += cwt[r]*kx; sy += cwt[r]*ky; sz += cwt[r]*kz;
    }
    x += (h/6.0f)*sx; y += (h/6.0f)*sy; z += (h/6.0f)*sz;

    // emit: xyzs[s] after every step; final xyz on last step
    if (lane < 16 && p < N){
      long long o = (long long)p * 3;
      out[N3 + (long long)s*N3 + o + 0] = x;
      out[N3 + (long long)s*N3 + o + 1] = y;
      out[N3 + (long long)s*N3 + o + 2] = z;
      if (s == Sc - 1){
        out[o + 0] = x; out[o + 1] = y; out[o + 2] = z;
      }
    }
  }
}

extern "C" void kernel_launch(void* const* d_in, const int* in_sizes, int n_in,
                              void* d_out, int out_size, void* d_ws, size_t ws_size,
                              hipStream_t stream) {
  const float* input = (const float*)d_in[0];
  const float* W1 = (const float*)d_in[1];
  const float* b1 = (const float*)d_in[2];
  const float* W2 = (const float*)d_in[3];
  const float* b2 = (const float*)d_in[4];
  const float* W4 = (const float*)d_in[5];
  const float* b4 = (const float*)d_in[6];
  const float* Wc = (const float*)d_in[7];
  const float* bc = (const float*)d_in[8];
  float* out = (float*)d_out;
  int N = in_sizes[0] / (Zc + 3);
  int blocks = (N + 127) / 128;
  warper_kernel<<<blocks, 256, LDS_BYTES, stream>>>(input, W1, b1, W2, b2, W4, b4, Wc, bc, out, N);
}
